// SubjectLayers_26645977104571
// MI455X (gfx1250) — matrix operand order repaired and checked by reading, not compiled, not run on hardware
//
#include <hip/hip_runtime.h>
#include <hip/hip_bf16.h>

// out[b,d,t] = sum_c x[b,c,t] * W[subj[b]][c,d]
// Batched GEMM: out_b = W_b^T (D x C) * X_b (C x T), all 1024, B=64.
// bf16x3 split-precision WMMA (hi*hi + hi*lo + lo*hi), f32 accumulate.
// Double-buffered LDS (one barrier/iter), bit-trick hi/lo split, b64 LDS stores,
// non-temporal output stores.

typedef __bf16 bf16;
typedef __attribute__((ext_vector_type(16))) __bf16 v16bf;
typedef __attribute__((ext_vector_type(8)))  __bf16 v8bf;
typedef __attribute__((ext_vector_type(8)))  float  v8f;

#define BATCH 64
#define CDIM 1024
#define DDIM 1024
#define TDIM 1024

#define BM 128          // d-tile per block
#define BN 128          // t-tile per block
#define BK 32           // k-step (== WMMA bf16 K)
#define LDSTR 40        // padded LDS row stride in bf16 (80B: 16B-aligned rows)
#define NKB (CDIM / BK)

static __device__ __forceinline__ v16bf cat8(v8bf a, v8bf b) {
  return __builtin_shufflevector(a, b, 0,1,2,3,4,5,6,7,8,9,10,11,12,13,14,15);
}

// Pack the high 16 bits (bf16 truncation) of two f32 bit-patterns into one dword.
static __device__ __forceinline__ unsigned pack_hi16(unsigned u0, unsigned u1) {
  return (u0 >> 16) | (u1 & 0xffff0000u);
}

// A-matrix 16x32 bf16 frag (ISA 7.12.2): lane half h: elems 0..7 <- K=8h.., 8..15 <- K=16+8h..
static __device__ __forceinline__ v16bf fragA(const bf16* rowp, int half) {
  v8bf a0 = *reinterpret_cast<const v8bf*>(rowp + 8 * half);
  v8bf a1 = *reinterpret_cast<const v8bf*>(rowp + 16 + 8 * half);
  return cat8(a0, a1);
}

// B-matrix 32x16 bf16 frag: lane half h: elems 0..7 <- K=8h.., 8..15 <- K=16+8h..
static __device__ __forceinline__ v16bf fragB(const bf16* rowp, int half) {
  v8bf b0 = *reinterpret_cast<const v8bf*>(rowp + 8 * half);
  v8bf b1 = *reinterpret_cast<const v8bf*>(rowp + 16 + 8 * half);
  return cat8(b0, b1);
}

__global__ __launch_bounds__(256)
void subject_layers_bf16x3(const float* __restrict__ x,
                           const long long* __restrict__ subjects,
                           const float* __restrict__ weights,
                           float* __restrict__ out) {
  // Double-buffered LDS tiles, [row][k] k-contiguous; hi/lo planes for
  // A (=W^T slice) and B (=X slice). 2 * 4 * 128*40*2B = 80 KB (<320 KB/WGP).
  __shared__ alignas(16) bf16 sAh[2][BM * LDSTR];
  __shared__ alignas(16) bf16 sAl[2][BM * LDSTR];
  __shared__ alignas(16) bf16 sBh[2][BN * LDSTR];
  __shared__ alignas(16) bf16 sBl[2][BN * LDSTR];

  const int tid = threadIdx.x;
  const int b  = blockIdx.z;
  const int d0 = blockIdx.y * BM;
  const int t0 = blockIdx.x * BN;

  const long long s = subjects[b];
  const float* wbase = weights + (size_t)s * CDIM * DDIM;  // [C][D] row-major
  const float* xbase = x + (size_t)b * CDIM * TDIM;        // [C][T] row-major

  // --- fill mapping: thread owns 4 consecutive k-rows x 4 columns per operand ---
  const int quad = tid & 31;   // 32 column-groups of 4 floats
  const int k4   = tid >> 5;   // 8 k-row quads (rows 4*k4 .. 4*k4+3)
  const int col  = 4 * quad;

  float4 stA[4];               // staging registers for next k-block
  float4 stB[4];

  auto gload = [&](int ks) {
#pragma unroll
    for (int rr = 0; rr < 4; ++rr) {
      const int k = ks + 4 * k4 + rr;
      stA[rr] = *reinterpret_cast<const float4*>(wbase + (size_t)k * DDIM + d0 + col);
      stB[rr] = *reinterpret_cast<const float4*>(xbase + (size_t)k * TDIM + t0 + col);
    }
  };

  auto lstore = [&](int buf) {
#pragma unroll
    for (int j = 0; j < 4; ++j) {
      const int rowi = col + j;
      const int off  = rowi * LDSTR + 4 * k4;   // 8B-aligned (byte = rowi*80 + 8*k4)
      // --- A plane ---
      {
        unsigned u0 = __builtin_bit_cast(unsigned, stA[0][j]);
        unsigned u1 = __builtin_bit_cast(unsigned, stA[1][j]);
        unsigned u2 = __builtin_bit_cast(unsigned, stA[2][j]);
        unsigned u3 = __builtin_bit_cast(unsigned, stA[3][j]);
        uint2 hv; hv.x = pack_hi16(u0, u1); hv.y = pack_hi16(u2, u3);
        *reinterpret_cast<uint2*>(&sAh[buf][off]) = hv;
        unsigned l0 = __builtin_bit_cast(unsigned, stA[0][j] - __builtin_bit_cast(float, u0 & 0xffff0000u));
        unsigned l1 = __builtin_bit_cast(unsigned, stA[1][j] - __builtin_bit_cast(float, u1 & 0xffff0000u));
        unsigned l2 = __builtin_bit_cast(unsigned, stA[2][j] - __builtin_bit_cast(float, u2 & 0xffff0000u));
        unsigned l3 = __builtin_bit_cast(unsigned, stA[3][j] - __builtin_bit_cast(float, u3 & 0xffff0000u));
        uint2 lv; lv.x = pack_hi16(l0, l1); lv.y = pack_hi16(l2, l3);
        *reinterpret_cast<uint2*>(&sAl[buf][off]) = lv;
      }
      // --- B plane ---
      {
        unsigned u0 = __builtin_bit_cast(unsigned, stB[0][j]);
        unsigned u1 = __builtin_bit_cast(unsigned, stB[1][j]);
        unsigned u2 = __builtin_bit_cast(unsigned, stB[2][j]);
        unsigned u3 = __builtin_bit_cast(unsigned, stB[3][j]);
        uint2 hv; hv.x = pack_hi16(u0, u1); hv.y = pack_hi16(u2, u3);
        *reinterpret_cast<uint2*>(&sBh[buf][off]) = hv;
        unsigned l0 = __builtin_bit_cast(unsigned, stB[0][j] - __builtin_bit_cast(float, u0 & 0xffff0000u));
        unsigned l1 = __builtin_bit_cast(unsigned, stB[1][j] - __builtin_bit_cast(float, u1 & 0xffff0000u));
        unsigned l2 = __builtin_bit_cast(unsigned, stB[2][j] - __builtin_bit_cast(float, u2 & 0xffff0000u));
        unsigned l3 = __builtin_bit_cast(unsigned, stB[3][j] - __builtin_bit_cast(float, u3 & 0xffff0000u));
        uint2 lv; lv.x = pack_hi16(l0, l1); lv.y = pack_hi16(l2, l3);
        *reinterpret_cast<uint2*>(&sBl[buf][off]) = lv;
      }
    }
  };

  // --- wave tiling: 8 waves = 4(M) x 2(N); wave tile 32(M) x 64(N) = 2x4 WMMA tiles ---
  const int wid = tid >> 5;
  const int wm = wid & 3;
  const int wn = wid >> 2;
  const int lanelo   = tid & 15;
  const int lanehalf = (tid >> 4) & 1;

  v8f acc[4][2];
#pragma unroll
  for (int ni = 0; ni < 4; ++ni)
#pragma unroll
    for (int mi = 0; mi < 2; ++mi) {
      v8f z = {0.f, 0.f, 0.f, 0.f, 0.f, 0.f, 0.f, 0.f};
      acc[ni][mi] = z;
    }

  // Prologue: fill buffer 0.
  gload(0);
  lstore(0);
  __syncthreads();

  for (int kb = 0; kb < NKB; ++kb) {
    const int cur = kb & 1;

    // Issue next k-block's global loads first: latency hides under WMMA work.
    if (kb + 1 < NKB) gload((kb + 1) * BK);

    // Compute from buffer `cur`: 24 v_wmma_f32_16x16x32_bf16 per wave.
    {
      const bf16* Ah = sAh[cur];
      const bf16* Al = sAl[cur];
      const bf16* Bh = sBh[cur];
      const bf16* Bl = sBl[cur];
      v16bf ah[2], al[2];
#pragma unroll
      for (int mi = 0; mi < 2; ++mi) {
        const int rowm = wm * 32 + mi * 16 + lanelo;
        ah[mi] = fragA(&Ah[rowm * LDSTR], lanehalf);
        al[mi] = fragA(&Al[rowm * LDSTR], lanehalf);
      }
#pragma unroll
      for (int ni = 0; ni < 4; ++ni) {
        const int rown = wn * 64 + ni * 16 + lanelo;
        v16bf bh = fragB(&Bh[rown * LDSTR], lanehalf);
        v16bf bl = fragB(&Bl[rown * LDSTR], lanehalf);
#pragma unroll
        for (int mi = 0; mi < 2; ++mi) {
          acc[ni][mi] = __builtin_amdgcn_wmma_f32_16x16x32_bf16(
              false, ah[mi], false, bh, (short)0, acc[ni][mi], false, false);
          acc[ni][mi] = __builtin_amdgcn_wmma_f32_16x16x32_bf16(
              false, ah[mi], false, bl, (short)0, acc[ni][mi], false, false);
          acc[ni][mi] = __builtin_amdgcn_wmma_f32_16x16x32_bf16(
              false, al[mi], false, bh, (short)0, acc[ni][mi], false, false);
        }
      }
    }

    // Store next k-block into the other buffer (its last readers finished at the
    // barrier that ended iteration kb-1), then one barrier publishes it.
    if (kb + 1 < NKB) lstore(cur ^ 1);
    __syncthreads();
  }

  // Epilogue: C/D 16x16 f32 layout -> out[b][d][t]; streaming (non-temporal) stores
  // so the output doesn't evict x/W tiles being reused through L2.
#pragma unroll
  for (int ni = 0; ni < 4; ++ni) {
    const int t = t0 + wn * 64 + ni * 16 + lanelo;
#pragma unroll
    for (int mi = 0; mi < 2; ++mi) {
      const int dbase = d0 + wm * 32 + mi * 16 + 8 * lanehalf;
      float* op = out + ((size_t)b * DDIM + dbase) * TDIM + t;
      v8f a = acc[ni][mi];
#pragma unroll
      for (int r = 0; r < 8; ++r)
        __builtin_nontemporal_store(a[r], op + (size_t)r * TDIM);
    }
  }
}

extern "C" void kernel_launch(void* const* d_in, const int* in_sizes, int n_in,
                              void* d_out, int out_size, void* d_ws, size_t ws_size,
                              hipStream_t stream) {
  (void)in_sizes; (void)n_in; (void)out_size; (void)d_ws; (void)ws_size;
  const float*     x        = (const float*)d_in[0];
  const long long* subjects = (const long long*)d_in[1];
  const float*     weights  = (const float*)d_in[2];
  float*           out      = (float*)d_out;

  dim3 grid(TDIM / BN, DDIM / BM, BATCH);
  subject_layers_bf16x3<<<grid, dim3(256), 0, stream>>>(x, subjects, weights, out);
}
